// net2_46591805227319
// MI455X (gfx1250) — matrix operand order repaired and checked by reading, not compiled, not run on hardware
//
#include <hip/hip_runtime.h>
#include <math.h>

// Shapes (fixed by the reference)
#define TT 512
#define BB 64
#define II 256
#define HH 1024
#define OO 256
#define NBLK 16     // persistent workgroups in the scan (64-col slice of H each)
#define NTHR 256    // 8 wave32s
#define PBLK 512    // blocks for the input-GEMM pre-pass (4 M-tiles each)

typedef unsigned short u16;
typedef __attribute__((ext_vector_type(8)))  unsigned short v8u;
typedef __attribute__((ext_vector_type(16))) unsigned short v16u;
typedef __attribute__((ext_vector_type(16))) __bf16        v16bf;
typedef __attribute__((ext_vector_type(8)))  float          v8f;

__device__ __host__ inline u16 f2bf(float f) {
  unsigned u = __builtin_bit_cast(unsigned, f);
  unsigned r = u + 0x7FFFu + ((u >> 16) & 1u);   // round-to-nearest-even
  return (u16)(r >> 16);
}

__device__ inline v16bf as_bf(v16u u) { return __builtin_bit_cast(v16bf, u); }

__device__ inline v16bf pack_ab(v8u lo, v8u hi) {
  v16u u;
#pragma unroll
  for (int e = 0; e < 8; ++e) { u[e] = lo[e]; u[e + 8] = hi[e]; }
  return __builtin_bit_cast(v16bf, u);
}

__device__ inline v8f wmma_bf16(v16bf a, v16bf b, v8f c) {
  // D(f32,16x16) = A(bf16,16x32) * B(bf16,32x16) + C
  return __builtin_amdgcn_wmma_f32_16x16x32_bf16(false, a, false, b, (short)0, c,
                                                 false, false);
}

// ---- A fragment (16x32 bf16, MxK) from row-major bf16, row stride ld.
// lanes<16: K=k0..k0+15; lanes>=16: K=k0+16..k0+31 (contiguous, matches load_b).
__device__ inline v16bf load_a(const u16* base, int m, int ld, int k0, int khalf) {
  const u16* p = base + (size_t)m * ld + k0 + khalf * 16;
  v8u lo = *(const v8u*)(p);
  v8u hi = *(const v8u*)(p + 8);
  return pack_ab(lo, hi);
}

// ---- B fragment (32x16 bf16, KxN) from Bt[n][k] row (contiguous K).
// lanes<16: K=k0..k0+15; lanes>=16: K=k0+16..k0+31.  One contiguous 32B load.
__device__ inline v16bf load_b(const u16* bt_row, int k0, int khalf) {
  return as_bf(*(const v16u*)(bt_row + k0 + khalf * 16));
}

// -------------------- init: convert weights to bf16, seed h, reset barrier ----------
__global__ void rnn_init_kernel(const float* __restrict__ wI, const float* __restrict__ wR,
                                const float* __restrict__ wO, const float* __restrict__ h0,
                                u16* wRt, u16* wIb, u16* wOb, u16* hbf, unsigned* bar) {
  int idx = blockIdx.x * blockDim.x + threadIdx.x;
  if (idx < HH * HH) {                 // wRt[n][k] = bf16(wR[k][n])  (pre-transpose)
    int n = idx / HH, k = idx % HH;
    wRt[idx] = f2bf(wR[(size_t)k * HH + n]);
  }
  if (idx < HH * II) wIb[idx] = f2bf(wI[idx]);      // wI[j][i], contiguous in i = K
  if (idx < OO * HH) wOb[idx] = f2bf(wO[idx]);      // wO[n][j], contiguous in j = K
  if (idx < BB * HH) hbf[idx] = f2bf(h0[idx % HH]); // broadcast h0 into buffer 0
  if (idx == 0) *bar = 0u;
}

// -------------------- pre-pass: P[t,b,:] = x[t,b,:] @ wI^T + bR  --------------------
// Fully parallel blocked GEMM.  Each block: 64 rows of [T*B, I] staged once in LDS
// as bf16; 8 waves x 8 N-tiles x 4 M-tiles, B fragments hoisted to registers.
// Result goes into out_h ([T*B, H]); scan step t reads its tile and overwrites it.
__global__ void __launch_bounds__(NTHR)
rnn_inproj_kernel(const float* __restrict__ x, const float* __restrict__ bR,
                  const u16* __restrict__ wIb, float* __restrict__ out_h) {
  __shared__ __align__(16) u16 xs[BB * (II + 8)];
  const int XLD = II + 8;

  const int tid = threadIdx.x;
  const int wv = tid >> 5, lane = tid & 31;
  const int lanelo = lane & 15, khalf = lane >> 4;
  const int row0 = blockIdx.x * 64;            // first of this block's 64 rows

  // stage 64x256 fp32 -> bf16 LDS (x read exactly once across the whole pre-pass)
  {
    const float4* xt4 = (const float4*)(x + (size_t)row0 * II);
#pragma unroll
    for (int it = 0; it < 16; ++it) {
      int i4 = tid + it * 256;                 // 4096 float4s
      float4 v = xt4[i4];
      int b = i4 >> 6, i = (i4 & 63) * 4;
      unsigned p0 = (unsigned)f2bf(v.x) | ((unsigned)f2bf(v.y) << 16);
      unsigned p1 = (unsigned)f2bf(v.z) | ((unsigned)f2bf(v.w) << 16);
      *(uint2*)(xs + b * XLD + i) = make_uint2(p0, p1);
    }
  }
  __syncthreads();

  for (int j = 0; j < 8; ++j) {
    const int nt = wv * 8 + j;                 // this wave's N-tile (0..63)
    const int col = nt * 16 + lanelo;
    const float bias = bR[col];
    const u16* wIrow = wIb + (size_t)col * II;

    v16bf bfr[8];                              // hoist all K fragments of B
#pragma unroll
    for (int kk = 0; kk < 8; ++kk) bfr[kk] = load_b(wIrow, kk * 32, khalf);

#pragma unroll
    for (int mt = 0; mt < 4; ++mt) {
      v8f acc;
#pragma unroll
      for (int r = 0; r < 8; ++r) acc[r] = bias;
#pragma unroll
      for (int kk = 0; kk < 8; ++kk) {
        v16bf a = load_a(xs, mt * 16 + lanelo, XLD, kk * 32, khalf);
        acc = wmma_bf16(a, bfr[kk], acc);
      }
#pragma unroll
      for (int r = 0; r < 8; ++r) {
        int row = row0 + mt * 16 + r + (khalf << 3);
        out_h[(size_t)row * HH + col] = acc[r];
      }
    }
  }
}

// -------------------- device-wide step barrier --------------------------------------
__device__ inline void grid_barrier(unsigned* bar, unsigned target) {
  __syncthreads();
  if (threadIdx.x == 0) {
    __threadfence();
    atomicAdd(bar, 1u);
    __builtin_amdgcn_s_cluster_barrier();   // NOP unless cluster-dispatched (ISA 6.6)
    while (__hip_atomic_load(bar, __ATOMIC_RELAXED, __HIP_MEMORY_SCOPE_AGENT) < target)
      __builtin_amdgcn_s_sleep(2);
    __threadfence();
  }
  __syncthreads();
}

// -------------------- main persistent RNN scan --------------------------------------
__global__ void __launch_bounds__(NTHR, 1)
rnn_scan_kernel(const float* __restrict__ bO, const u16* __restrict__ wRt,
                const u16* __restrict__ wOb, u16* hbf, float* __restrict__ out_o,
                float* __restrict__ out_h, unsigned* bar) {
  __shared__ __align__(16) float osum[4 * 32 * 8];   // K-split partials for output GEMM

  const int tid  = threadIdx.x;
  const int blk  = blockIdx.x;          // owns H columns [blk*64, blk*64+64)
  const int wv   = tid >> 5;
  const int lane = tid & 31;
  const int lanelo = lane & 15;
  const int khalf  = lane >> 4;

  const int nt  = wv >> 1;              // n-tile within slice (0..3)
  const int mtA = (wv & 1) * 2;         // two m-tiles sharing the B fragment
  const int mtB = mtA + 1;
  const int n0  = blk * 64 + nt * 16;
  const int ncol = n0 + lanelo;         // this lane's wR column (absolute j)

  const int n0o   = blk * 16;           // 16-wide O-column slice of this block
  const int ncolO = n0o + lanelo;
  const float biasO = bO[ncolO];
  const int omt = wv & 3;               // output tile / K-half split across 8 waves
  const int oks = (wv >> 2) * (HH / 2);

  const u16* wRrow = wRt + (size_t)ncol * HH;
  const u16* wOrow = wOb + (size_t)ncolO * HH;

  for (int t = 0; t < TT; ++t) {
    u16* hcur = hbf + (size_t)(t & 1) * (BB * HH);
    u16* hnxt = hbf + (size_t)((t + 1) & 1) * (BB * HH);

    // ---- acc init: precomputed P[t] = x_t@wI^T + bR (written by pre-pass) ----
    v8f acc0, acc1, acc0h, acc1h;
    {
      const float* P = out_h + (size_t)t * BB * HH;
#pragma unroll
      for (int r = 0; r < 8; ++r) {
        acc0[r]  = P[(size_t)(mtA * 16 + r + (khalf << 3)) * HH + ncol];
        acc1[r]  = P[(size_t)(mtB * 16 + r + (khalf << 3)) * HH + ncol];
        acc0h[r] = 0.f;
        acc1h[r] = 0.f;
      }
      if (t + 1 < TT)   // pull next step's P tile toward L2/WGP$
        __builtin_prefetch(out_h + (size_t)((t + 1) * BB + mtA * 16 + lanelo) * HH + n0, 0, 1);
    }

    // ---- recurrent GEMM: 4 independent WMMA chains of length 16 per wave ----
#pragma unroll 2
    for (int k0 = 0; k0 < HH; k0 += 64) {
      v16bf blo = load_b(wRrow, k0, khalf);
      v16bf bhi = load_b(wRrow, k0 + 32, khalf);
      v16bf a0l = load_a(hcur, mtA * 16 + lanelo, HH, k0, khalf);
      v16bf a0h = load_a(hcur, mtA * 16 + lanelo, HH, k0 + 32, khalf);
      v16bf a1l = load_a(hcur, mtB * 16 + lanelo, HH, k0, khalf);
      v16bf a1h = load_a(hcur, mtB * 16 + lanelo, HH, k0 + 32, khalf);
      acc0  = wmma_bf16(a0l, blo, acc0);
      acc0h = wmma_bf16(a0h, bhi, acc0h);
      acc1  = wmma_bf16(a1l, blo, acc1);
      acc1h = wmma_bf16(a1h, bhi, acc1h);
    }

    // ---- tanh, write hidden_states (fp32, overwrites P[t]) and next-h (bf16) ----
#pragma unroll
    for (int r = 0; r < 8; ++r) {
      int Ma = mtA * 16 + r + (khalf << 3);
      int Mb = mtB * 16 + r + (khalf << 3);
      float va = tanhf(acc0[r] + acc0h[r]);
      float vb = tanhf(acc1[r] + acc1h[r]);
      out_h[(size_t)(t * BB + Ma) * HH + n0 + lanelo] = va;
      out_h[(size_t)(t * BB + Mb) * HH + n0 + lanelo] = vb;
      hnxt[Ma * HH + n0 + lanelo] = f2bf(va);
      hnxt[Mb * HH + n0 + lanelo] = f2bf(vb);
    }

    grid_barrier(bar, (unsigned)(NBLK * (t + 1)));   // h_new visible device-wide

    // ---- output GEMM: o = h_new @ wO^T + bO.  8 waves: 4 tiles x 2 K-halves ----
    v8f acco;
#pragma unroll
    for (int r = 0; r < 8; ++r) acco[r] = 0.f;
#pragma unroll 4
    for (int k0 = 0; k0 < HH / 2; k0 += 32) {
      v16bf bf = load_b(wOrow, oks + k0, khalf);
      v16bf a0 = load_a(hnxt, omt * 16 + lanelo, HH, oks + k0, khalf);
      acco = wmma_bf16(a0, bf, acco);
    }
    if (wv >= 4) {                       // upper-K partials through LDS
#pragma unroll
      for (int r = 0; r < 8; ++r) osum[(wv - 4) * 256 + lane * 8 + r] = acco[r];
    }
    __syncthreads();
    if (wv < 4) {
#pragma unroll
      for (int r = 0; r < 8; ++r) {
        int M = omt * 16 + r + (khalf << 3);
        float v = acco[r] + osum[wv * 256 + lane * 8 + r] + biasO;
        out_o[(size_t)(t * BB + M) * OO + ncolO] = v;
      }
    }
  }
}

// -------------------- host entry ----------------------------------------------------
extern "C" void kernel_launch(void* const* d_in, const int* in_sizes, int n_in,
                              void* d_out, int out_size, void* d_ws, size_t ws_size,
                              hipStream_t stream) {
  const float* x  = (const float*)d_in[0];
  const float* h0 = (const float*)d_in[1];
  const float* wI = (const float*)d_in[2];
  const float* wR = (const float*)d_in[3];
  const float* wO = (const float*)d_in[4];
  const float* bR = (const float*)d_in[5];
  const float* bO = (const float*)d_in[6];

  float* out_o = (float*)d_out;                       // [T,B,O]
  float* out_h = out_o + (size_t)TT * BB * OO;        // [T,B,H] (doubles as P buffer)

  // workspace (~3.3 MB): bf16 weights (wR transposed), ping-pong bf16 h, barrier word
  u16* wRt = (u16*)d_ws;                              // H*H
  u16* wIb = wRt + (size_t)HH * HH;                   // H*I
  u16* wOb = wIb + (size_t)HH * II;                   // O*H
  u16* hbf = wOb + (size_t)OO * HH;                   // 2*B*H
  unsigned* bar = (unsigned*)(hbf + (size_t)2 * BB * HH);

  rnn_init_kernel<<<(HH * HH + NTHR - 1) / NTHR, NTHR, 0, stream>>>(
      wI, wR, wO, h0, wRt, wIb, wOb, hbf, bar);

  rnn_inproj_kernel<<<PBLK, NTHR, 0, stream>>>(x, bR, wIb, out_h);

  rnn_scan_kernel<<<NBLK, NTHR, 0, stream>>>(
      bO, wRt, wOb, hbf, out_o, out_h, bar);
}
